// DynamicSeekerAdapter_76991583748287
// MI455X (gfx1250) — matrix operand order repaired and checked by reading, not compiled, not run on hardware
//
#include <hip/hip_runtime.h>
#include <hip/hip_bf16.h>
#include <math.h>

// ---------------------------------------------------------------------------
// DynamicSeekerAdapter for MI455X (gfx1250, wave32).
//
// Roofline: dominated by reading image_features (256MB) and writing out
// (256MB) -> ~23us @ 23.3 TB/s.  The 8.6 GFLOP downproject GEMM runs as
// v_wmma_f32_16x16x32_f16 and is fused with the "out = image + gamma*bu"
// baseline write so image_features is read from HBM exactly once.
// ---------------------------------------------------------------------------

typedef _Float16 v16h __attribute__((ext_vector_type(16)));
typedef float    v8f  __attribute__((ext_vector_type(8)));

#define BB   4
#define NN   16384
#define CC   1024
#define DD   64
#define MM   16
#define KSEL 64
#define SS   80
#define HH   4
#define DH   16

__device__ __forceinline__ float gelu_exact(float x) {
    return 0.5f * x * (1.0f + erff(x * 0.70710678118654752f));
}

// ---------------------------------------------------------------------------
// Kernel 1: normalized text selector vector per batch:  seln[b][d]
// ---------------------------------------------------------------------------
__global__ void k_seln(const float* __restrict__ text, float* __restrict__ seln) {
    __shared__ float sh[256];
    __shared__ float nrm[BB];
    int t = threadIdx.x;          // 256 threads = 4 batches * 64 dims
    int b = t >> 6, d = t & 63;
    float v = text[(b * 20 + 0) * 512 + d];   // text_features[b, 0, 0:64]
    sh[t] = v * v;
    __syncthreads();
    if (d == 0) {
        float s = 0.f;
        for (int i = 0; i < 64; ++i) s += sh[b * 64 + i];
        nrm[b] = sqrtf(s);
    }
    __syncthreads();
    seln[t] = v / fmaxf(nrm[b], 1e-12f);
}

// ---------------------------------------------------------------------------
// Kernel 2: fused  a = gelu(X@Wd + bd)  via WMMA f16 ->f32,
//           scores[row] = (a_row . seln_b) / max(||a_row||, 1e-12),
//           out = X + gamma*bu   (baseline for the sparse delta)
// Block = 256 threads (8 waves), 128 rows per block, K-slabs of 32.
// ---------------------------------------------------------------------------
__global__ void __launch_bounds__(256) k_fused_gemm(
    const float* __restrict__ X,      // (B*N, C)
    const float* __restrict__ Wd,     // (C, D)
    const float* __restrict__ bd,     // (D)
    const float* __restrict__ bu,     // (C)
    const float* __restrict__ seln,   // (B, D)
    const float* __restrict__ gammap, // (1)
    float* __restrict__ a_out,        // (B*N, D)
    float* __restrict__ scores,       // (B*N)
    float* __restrict__ out)          // (B*N, C)
{
    __shared__ __align__(16) _Float16 AS[128 * 32];   // X slab, f16
    __shared__ __align__(16) _Float16 BS[64 * 32];    // Wd slab transposed: BS[n*32+k]

    const int tid  = threadIdx.x;
    const int lane = tid & 31;
    const int wave = tid >> 5;                 // 0..7, 16 rows each
    const int rowBase = blockIdx.x * 128;
    const int bidx = rowBase / NN;             // 128 | 16384 -> single batch per block
    const float gamma = gammap[0];

    const int rq = tid >> 3;                   // 0..31 : row group for staging
    const int c8 = tid & 7;                    // 0..7  : float4 column slot

    v8f acc[4];
    for (int j = 0; j < 4; ++j)
        for (int i = 0; i < 8; ++i) acc[j][i] = 0.0f;

    const int half = lane >> 4;                // 0 | 1
    const int l16  = lane & 15;

    for (int kb = 0; kb < CC; kb += 32) {
        // ---- stage Wd slab transposed (k contiguous per output column n) ----
        for (int i = 0; i < 8; ++i) {
            int e  = i * 256 + tid;            // 0..2047
            int kl = e >> 6, n = e & 63;
            BS[n * 32 + kl] = (_Float16)Wd[(kb + kl) * DD + n];
        }
        // ---- stage X slab (f16) and fuse the baseline output write ----
        float4 bu4 = *(const float4*)(bu + kb + c8 * 4);
        bu4.x *= gamma; bu4.y *= gamma; bu4.z *= gamma; bu4.w *= gamma;
        for (int i = 0; i < 4; ++i) {
            int rl = rq + i * 32;
            size_t g = (size_t)(rowBase + rl) * CC + kb + c8 * 4;
            float4 xv = *(const float4*)(X + g);
            float4 ov;
            ov.x = xv.x + bu4.x; ov.y = xv.y + bu4.y;
            ov.z = xv.z + bu4.z; ov.w = xv.w + bu4.w;
            *(float4*)(out + g) = ov;          // out = image + gamma*bu
            _Float16* ap = &AS[rl * 32 + c8 * 4];
            ap[0] = (_Float16)xv.x; ap[1] = (_Float16)xv.y;
            ap[2] = (_Float16)xv.z; ap[3] = (_Float16)xv.w;
        }
        if (kb + 32 < CC)                       // global_prefetch_b8 next slab
            __builtin_prefetch(X + (size_t)(rowBase + rq) * CC + kb + 32 + c8 * 4, 0, 1);
        __syncthreads();

        // ---- fragments + WMMA ----
        union Fr { v16h v; uint4 q[2]; };
        Fr fa;
        {   // A 16x32 f16: lane half h -> K = 8h..8h+7 and 8h+16..8h+23
            const _Float16* p = &AS[(wave * 16 + l16) * 32 + half * 8];
            fa.q[0] = *(const uint4*)p;
            fa.q[1] = *(const uint4*)(p + 16);
        }
        for (int j = 0; j < 4; ++j) {
            Fr fb;
            {   // B 32x16 f16: lane half h holds K = 8h..8h+7 and 8h+16..8h+23 of column n
                const _Float16* p = &BS[(j * 16 + l16) * 32 + half * 8];
                fb.q[0] = *(const uint4*)p;
                fb.q[1] = *(const uint4*)(p + 16);
            }
            acc[j] = __builtin_amdgcn_wmma_f32_16x16x32_f16(
                false, fa.v, false, fb.v, (short)0, acc[j], false, false);
        }
        __syncthreads();
    }

    // ---- epilogue: bias + gelu, store a, fused row-norm + selector dot ----
    for (int i = 0; i < 8; ++i) {
        size_t row = (size_t)rowBase + wave * 16 + half * 8 + i;
        float ss = 0.f, dd = 0.f;
        for (int j = 0; j < 4; ++j) {
            int n = j * 16 + l16;
            float g = gelu_exact(acc[j][i] + bd[n]);
            a_out[row * DD + n] = g;
            ss += g * g;
            dd += g * seln[bidx * DD + n];
        }
        // reduce across the 16 lanes owning this row (wave32-safe xor masks)
        for (int m = 8; m >= 1; m >>= 1) {
            ss += __shfl_xor(ss, m, 32);
            dd += __shfl_xor(dd, m, 32);
        }
        if (l16 == 0) scores[row] = dd / fmaxf(sqrtf(ss), 1e-12f);
    }
}

// ---------------------------------------------------------------------------
// Kernel 3: top-64 per batch (iterative block argmax over L2-hot scores),
//           then sort ascending.  scores is scratch and recomputed each call.
// ---------------------------------------------------------------------------
__global__ void k_topk(float* __restrict__ scores, int* __restrict__ idx) {
    __shared__ float sv[256];
    __shared__ int   si[256];
    __shared__ int   sel[KSEL];
    int b = blockIdx.x, t = threadIdx.x;
    float* sc = scores + b * NN;
    for (int p = 0; p < KSEL; ++p) {
        float best = -__builtin_inff(); int bi = 0;
        for (int n = t; n < NN; n += 256) {
            float v = sc[n];
            if (v > best) { best = v; bi = n; }
        }
        sv[t] = best; si[t] = bi;
        __syncthreads();
        for (int s = 128; s > 0; s >>= 1) {
            if (t < s) {
                if (sv[t + s] > sv[t] ||
                    (sv[t + s] == sv[t] && si[t + s] < si[t])) {
                    sv[t] = sv[t + s]; si[t] = si[t + s];
                }
            }
            __syncthreads();
        }
        if (t == 0) { sel[p] = si[0]; sc[si[0]] = -__builtin_inff(); }
        __syncthreads();
    }
    if (t == 0) {                                   // sort 64 ascending
        for (int i = 1; i < KSEL; ++i) {
            int key = sel[i], j = i - 1;
            while (j >= 0 && sel[j] > key) { sel[j + 1] = sel[j]; --j; }
            sel[j + 1] = key;
        }
    }
    __syncthreads();
    if (t < KSEL) idx[b * KSEL + t] = sel[t];
}

// ---------------------------------------------------------------------------
// Kernel 4: per batch -- gather, layernorm, 4-head attention over S=80 tokens,
//           output projection, enh_sparse = comb + attn_out (rows 16..79).
// Dynamic LDS: x / q / k / v, 80*64 f32 each = 80KB.
// ---------------------------------------------------------------------------
__global__ void __launch_bounds__(256) k_attn(
    const float* __restrict__ a, const int* __restrict__ idx,
    const float* __restrict__ mq,
    const float* __restrict__ Wq, const float* __restrict__ bq,
    const float* __restrict__ Wk, const float* __restrict__ bk,
    const float* __restrict__ Wv, const float* __restrict__ bv,
    const float* __restrict__ Wo, const float* __restrict__ bo,
    const float* __restrict__ lnw, const float* __restrict__ lnb,
    float* __restrict__ enh_sp)
{
    extern __shared__ float sm[];
    float* xs = sm;                 // 5120 floats
    float* qs = sm + 5120;
    float* ks = sm + 10240;
    float* vs = sm + 15360;
    int b = blockIdx.x, t = threadIdx.x;

    // layernorm(comb) -> xs
    if (t < SS) {
        int s = t;
        const float* src = (s < MM)
            ? (mq + s * DD)
            : (a + ((size_t)b * NN + idx[b * KSEL + (s - MM)]) * DD);
        float mu = 0.f;
        for (int d = 0; d < DD; ++d) mu += src[d];
        mu *= (1.0f / DD);
        float var = 0.f;
        for (int d = 0; d < DD; ++d) { float z = src[d] - mu; var += z * z; }
        var *= (1.0f / DD);
        float rs = rsqrtf(var + 1e-5f);
        for (int d = 0; d < DD; ++d)
            xs[s * DD + d] = (src[d] - mu) * rs * lnw[d] + lnb[d];
    }
    __syncthreads();

    // q, k, v projections
    for (int e = t; e < SS * DD; e += 256) {
        int s = e / DD, n = e % DD;
        float aq = bq[n], ak = bk[n], av = bv[n];
        for (int kd = 0; kd < DD; ++kd) {
            float xv = xs[s * DD + kd];
            aq += xv * Wq[kd * DD + n];
            ak += xv * Wk[kd * DD + n];
            av += xv * Wv[kd * DD + n];
        }
        qs[e] = aq; ks[e] = ak; vs[e] = av;
    }
    __syncthreads();

    // attention per (s, h) with two-pass online softmax; o overwrites xs
    for (int p = t; p < SS * HH; p += 256) {
        int s = p >> 2, h = p & 3;
        float qh[DH], o[DH];
        for (int d = 0; d < DH; ++d) { qh[d] = qs[s * DD + h * DH + d]; o[d] = 0.f; }
        const float scale = 0.25f;                  // 1/sqrt(16)
        float mx = -__builtin_inff();
        for (int tt = 0; tt < SS; ++tt) {
            float dp = 0.f;
            for (int d = 0; d < DH; ++d) dp += qh[d] * ks[tt * DD + h * DH + d];
            mx = fmaxf(mx, dp * scale);
        }
        float se = 0.f;
        for (int tt = 0; tt < SS; ++tt) {
            float dp = 0.f;
            for (int d = 0; d < DH; ++d) dp += qh[d] * ks[tt * DD + h * DH + d];
            float w = __expf(dp * scale - mx);
            se += w;
            for (int d = 0; d < DH; ++d) o[d] += w * vs[tt * DD + h * DH + d];
        }
        float inv = 1.0f / se;
        for (int d = 0; d < DH; ++d) xs[s * DD + h * DH + d] = o[d] * inv;
    }
    __syncthreads();

    // enh_sparse = comb_sparse + (o @ Wo + bo), sparse rows only
    for (int e = t; e < KSEL * DD; e += 256) {
        int r = e / DD, n = e % DD;
        int s = r + MM;
        float dsum = bo[n];
        for (int kd = 0; kd < DD; ++kd) dsum += xs[s * DD + kd] * Wo[kd * DD + n];
        size_t rr = (size_t)b * NN + idx[b * KSEL + r];
        enh_sp[(b * KSEL + r) * DD + n] = a[rr * DD + n] + dsum;
    }
}

// ---------------------------------------------------------------------------
// Kernel 5: sparse fix-up:  out[b, idx[r], :] += gamma * (enh_row @ Wu)
// (out already holds image + gamma*bu everywhere)
// ---------------------------------------------------------------------------
__global__ void k_scatter(const float* __restrict__ enh_sp,
                          const int* __restrict__ idx,
                          const float* __restrict__ Wu,
                          const float* __restrict__ gammap,
                          float* __restrict__ out)
{
    __shared__ float er[DD];
    int blk = blockIdx.x;
    int b = blk >> 6, r = blk & 63;
    int t = threadIdx.x;
    if (t < DD) er[t] = enh_sp[(b * KSEL + r) * DD + t];
    __syncthreads();
    float gamma = gammap[0];
    size_t row = (size_t)b * NN + idx[b * KSEL + r];
    for (int c = t; c < CC; c += 256) {
        float acc = 0.f;
        for (int kd = 0; kd < DD; ++kd) acc += er[kd] * Wu[kd * CC + c];
        out[row * CC + c] += gamma * acc;
    }
}

// ---------------------------------------------------------------------------
extern "C" void kernel_launch(void* const* d_in, const int* in_sizes, int n_in,
                              void* d_out, int out_size, void* d_ws, size_t ws_size,
                              hipStream_t stream) {
    (void)in_sizes; (void)n_in; (void)out_size; (void)ws_size;
    const float* image = (const float*)d_in[0];
    const float* text  = (const float*)d_in[1];
    const float* Wd    = (const float*)d_in[2];
    const float* bd    = (const float*)d_in[3];
    const float* Wu    = (const float*)d_in[4];
    const float* bu    = (const float*)d_in[5];
    const float* mq    = (const float*)d_in[6];
    const float* Wq    = (const float*)d_in[7];
    const float* bq    = (const float*)d_in[8];
    const float* Wk    = (const float*)d_in[9];
    const float* bk    = (const float*)d_in[10];
    const float* Wv    = (const float*)d_in[11];
    const float* bv    = (const float*)d_in[12];
    const float* Wo    = (const float*)d_in[13];
    const float* bo    = (const float*)d_in[14];
    const float* lnw   = (const float*)d_in[15];
    const float* lnb   = (const float*)d_in[16];
    const float* gamma = (const float*)d_in[17];
    float* out = (float*)d_out;

    // workspace layout (floats): seln | a | scores | idx(int) | enh_sparse
    float* ws     = (float*)d_ws;
    float* seln   = ws;                                    // 256
    float* a      = ws + 256;                              // 4,194,304
    float* scores = a + (size_t)BB * NN * DD;              // 65,536
    int*   idx    = (int*)(scores + (size_t)BB * NN);      // 256
    float* enh    = (float*)(idx + 256);                   // 16,384

    k_seln<<<1, 256, 0, stream>>>(text, seln);
    k_fused_gemm<<<(BB * NN) / 128, 256, 0, stream>>>(
        image, Wd, bd, bu, seln, gamma, a, scores, out);
    k_topk<<<BB, 256, 0, stream>>>(scores, idx);
    k_attn<<<BB, 256, 81920, stream>>>(a, idx, mq, Wq, bq, Wk, bk,
                                       Wv, bv, Wo, bo, lnw, lnb, enh);
    k_scatter<<<BB * KSEL, 256, 0, stream>>>(enh, idx, Wu, gamma, out);
}
